// AdaptiveRecursiveReasoner_37864431681813
// MI455X (gfx1250) — compile-verified, matrix operand order checked by reading, not run on hardware
//
#include <hip/hip_runtime.h>
#include <hip/hip_bf16.h>
#include <math.h>

// ---------------- problem constants (match reference) ----------------
#define B_    4096
#define H_    2048
#define CH1_  256
#define CH2_  128
#define NMEM_ 4096
#define MAXD  5
#define MIND  1
#define CONF_TH 0.85f
#define SIM_TH  0.9f

// async global->LDS staging of the pre-packed B tiles (ASYNCcnt path)
#define USE_ASYNC_B 1

typedef __attribute__((ext_vector_type(16))) __bf16 v16bf;
typedef __attribute__((ext_vector_type(8)))  __bf16 v8bf;
typedef __attribute__((ext_vector_type(8)))  float  v8f;

// ---------------- GEMM tiling ----------------
#define BM 128
#define BN 128
#define BK 32
// fragment-ordered LDS tiles (per ISA 16-bit layout):
//   lane = (m|n & 15) + 16*((k>>3)&1), e = (k&7) | ((k&16)>>1)
//   AsF[g=0..7][lane][e]  (8*512 bf16 = 8KB)   g = m>>4
//   BsF[t=0..7][lane][e]  (8*512 bf16 = 8KB)   t = n>>4

enum { ACT_NONE = 0, ACT_GELU = 1 };

struct Ctrl {
  float conf_sum;
  float conf_mean;
  int   amax_depth;
  int   depth;
  int   stopped;
  int   do_compute;
  int   pending_stop;
};

__device__ __forceinline__ float gelu_f(float x) {
  const float k0 = 0.7978845608028654f, k1 = 0.044715f;
  return 0.5f * x * (1.f + tanhf(k0 * (x + k1 * x * x * x)));
}
__device__ __forceinline__ float sigm_f(float x) { return 1.f / (1.f + expf(-x)); }

// ---------------------------------------------------------------------------
// Pre-pack a weight matrix into bf16 WMMA-fragment order (once per launch).
// Octet o = ((kb*(N/64)+nb)*256 + t*64 + lane*2 + (e0>>3)); 64-col granular.
// TRANSB: source is [N,K] row-major (keys_n), else [K,N] row-major.
// ---------------------------------------------------------------------------
template<bool TRANSB>
__global__ __launch_bounds__(256)
void pack_b(const float* __restrict__ Bm, __bf16* __restrict__ Bp, int N, int K) {
  size_t o = (size_t)blockIdx.x * 256 + threadIdx.x;
  int e0h  = (int)(o & 1);
  int lane = (int)((o >> 1) & 31);
  int t    = (int)((o >> 6) & 3);
  size_t blk = o >> 8;
  int nblk = N >> 6;
  int nb = (int)(blk % nblk);
  int kb = (int)(blk / nblk);
  int n  = nb * 64 + t * 16 + (lane & 15);
  int kh = lane >> 4;
  int kbase = kb * 32 + kh * 8 + (e0h ? 16 : 0);
  v8bf r;
  #pragma unroll
  for (int e = 0; e < 8; ++e) {
    int k = kbase + e;
    float v = TRANSB ? Bm[(size_t)n * K + k] : Bm[(size_t)k * N + n];
    r[e] = (__bf16)v;
  }
  *(v8bf*)&Bp[o * 8] = r;
}

// ---------------------------------------------------------------------------
// Double-buffered bf16 WMMA GEMM.
// Block tile 128x128 (BK=32); 8 waves in 4x2: each wave = 32 rows x 64 cols
// = 2 A-frags x 4 B-frags = 8 WMMA per K-step.
// A staged via regs (f32->bf16 cvt); B staged async global->LDS (pure copy).
// ---------------------------------------------------------------------------
template<int ACT, bool SCALEN>
__global__ __launch_bounds__(256)
void gemm_bf16_wmma(const float* __restrict__ A, const __bf16* __restrict__ Bp,
                    const float* __restrict__ bias, const float* __restrict__ nscale,
                    float* __restrict__ C, int M, int N, int K) {
  __shared__ alignas(32) __bf16 AsF[2][BM * BK];   // 2 x 8KB
  __shared__ alignas(32) __bf16 BsF[2][BK * BN];   // 2 x 8KB

  const int tid  = threadIdx.x;
  const int lane = tid & 31;
  const int wv   = tid >> 5;
  const int m0g  = blockIdx.y * BM;
  const int n0g  = blockIdx.x * BN;
  const int mr   = lane & 15;
  const int kh   = lane >> 4;
  const int nblk = N >> 6;

  const int g0  = (wv & 3) * 2;   // A fragment groups g0, g0+1
  const int tb0 = (wv >> 2) * 4;  // B fragment tiles tb0..tb0+3

  // per-thread constant staging geometry (2 A octet-tasks)
  int am[2], aj[2], aoff[2];
  #pragma unroll
  for (int it = 0; it < 2; ++it) {
    int i = tid + it * 256;       // 0..511
    am[it] = i >> 2;
    aj[it] = i & 3;
    int g  = am[it] >> 4;
    int ln = (am[it] & 15) + ((aj[it] & 1) << 4);
    int e0 = (aj[it] & 2) << 2;
    aoff[it] = g * 512 + ln * 16 + e0;
  }

  float4 ra[2][2];
#if !USE_ASYNC_B
  v8bf rb[2];
#endif

  auto gloadA = [&](int kk) {
    #pragma unroll
    for (int it = 0; it < 2; ++it) {
      const float* ap = A + (size_t)(m0g + am[it]) * K + kk + aj[it] * 8;
      ra[it][0] = ((const float4*)ap)[0];
      ra[it][1] = ((const float4*)ap)[1];
    }
  };
  auto stageB = [&](int kk, int b) {
    const __bf16* bsrc = Bp + ((size_t)(kk >> 5) * nblk + (n0g >> 6)) * 2048;
#if USE_ASYNC_B
    unsigned l0 = (unsigned)(uintptr_t)&BsF[b][(size_t)tid * 8];
    unsigned l1 = (unsigned)(uintptr_t)&BsF[b][(size_t)(tid + 256) * 8];
    const __bf16* g0p = bsrc + (size_t)tid * 8;
    const __bf16* g1p = bsrc + (size_t)(tid + 256) * 8;
    asm volatile("global_load_async_to_lds_b128 %0, %1, off"
                 :: "v"(l0), "v"(g0p) : "memory");
    asm volatile("global_load_async_to_lds_b128 %0, %1, off"
                 :: "v"(l1), "v"(g1p) : "memory");
#else
    rb[0] = *(const v8bf*)&bsrc[(size_t)tid * 8];
    rb[1] = *(const v8bf*)&bsrc[(size_t)(tid + 256) * 8];
#endif
  };
  auto lstoreA = [&](int b) {
    #pragma unroll
    for (int it = 0; it < 2; ++it) {
      v8bf r;
      r[0] = (__bf16)ra[it][0].x; r[1] = (__bf16)ra[it][0].y;
      r[2] = (__bf16)ra[it][0].z; r[3] = (__bf16)ra[it][0].w;
      r[4] = (__bf16)ra[it][1].x; r[5] = (__bf16)ra[it][1].y;
      r[6] = (__bf16)ra[it][1].z; r[7] = (__bf16)ra[it][1].w;
      *(v8bf*)&AsF[b][aoff[it]] = r;
    }
#if !USE_ASYNC_B
    *(v8bf*)&BsF[b][(size_t)tid * 8]         = rb[0];
    *(v8bf*)&BsF[b][(size_t)(tid + 256) * 8] = rb[1];
#endif
  };
  auto wait_async = [&]() {
#if USE_ASYNC_B
    asm volatile("s_wait_asynccnt 0x0" ::: "memory");
#endif
  };

  v8f acc[2][4] = {{v8f{}, v8f{}, v8f{}, v8f{}}, {v8f{}, v8f{}, v8f{}, v8f{}}};

  stageB(0, 0);
  gloadA(0);
  lstoreA(0);
  wait_async();
  __syncthreads();

  int buf = 0;
  for (int kk = 0; kk < K; kk += BK) {
    const bool has_next = (kk + BK) < K;
    if (has_next) {
      stageB(kk + BK, buf ^ 1);   // async copy straight into LDS
      gloadA(kk + BK);            // hide latency under WMMA
    }
    if (kk + 2 * BK < K)          // global_prefetch_b8
      __builtin_prefetch(&A[(size_t)(m0g + (tid & 127)) * K + kk + 2 * BK], 0, 1);

    // ---- compute current buffer: 12 ds_load_b128, 8 WMMA ----
    const __bf16* Ab = AsF[buf];
    const __bf16* Bb = BsF[buf];
    const v16bf a0 = *(const v16bf*)&Ab[(g0    ) * 512 + lane * 16];
    const v16bf a1 = *(const v16bf*)&Ab[(g0 + 1) * 512 + lane * 16];
    #pragma unroll
    for (int t = 0; t < 4; ++t) {
      v16bf b = *(const v16bf*)&Bb[(tb0 + t) * 512 + lane * 16];
      acc[0][t] = __builtin_amdgcn_wmma_f32_16x16x32_bf16(
          false, a0, false, b, (short)0, acc[0][t], false, false);
      acc[1][t] = __builtin_amdgcn_wmma_f32_16x16x32_bf16(
          false, a1, false, b, (short)0, acc[1][t], false, false);
    }

    if (has_next) lstoreA(buf ^ 1);
    wait_async();                  // async B copy complete before publish
    __syncthreads();
    buf ^= 1;
  }

  // ---- epilogue ----
  const int rsel = kh * 8;
  #pragma unroll
  for (int ag = 0; ag < 2; ++ag) {
    #pragma unroll
    for (int t = 0; t < 4; ++t) {
      int gn = n0g + (wv >> 2) * 64 + t * 16 + mr;
      float bv = bias ? bias[gn] : 0.f;
      float sc = SCALEN ? nscale[gn] : 1.f;
      #pragma unroll
      for (int r = 0; r < 8; ++r) {
        int gm = m0g + (wv & 3) * 32 + ag * 16 + rsel + r;
        float v = acc[ag][t][r] + bv;
        if (SCALEN) v *= sc;
        if (ACT == ACT_GELU) v = gelu_f(v);
        C[(size_t)gm * N + gn] = v;
      }
    }
  }
}

// ---------------------------------------------------------------------------
__global__ __launch_bounds__(256)
void normalize_rows(const float* __restrict__ in, float* __restrict__ out, int ncols) {
  __shared__ float red[8];
  __shared__ float s_inv;
  const int row = blockIdx.x, tid = threadIdx.x;
  float ss = 0.f;
  for (int c = tid; c < ncols; c += 256) {
    float v = in[(size_t)row * ncols + c];
    ss += v * v;
  }
  #pragma unroll
  for (int off = 16; off; off >>= 1) ss += __shfl_xor(ss, off, 32);
  if ((tid & 31) == 0) red[tid >> 5] = ss;
  __syncthreads();
  if (tid == 0) {
    float t = 0.f;
    #pragma unroll
    for (int i = 0; i < 8; ++i) t += red[i];
    s_inv = 1.f / fmaxf(sqrtf(t), 1e-8f);
  }
  __syncthreads();
  float inv = s_inv;
  for (int c = tid; c < ncols; c += 256)
    out[(size_t)row * ncols + c] = in[(size_t)row * ncols + c] * inv;
}

__global__ __launch_bounds__(256)
void conf_head(const float* __restrict__ t2, const float* __restrict__ cw3,
               const float* __restrict__ cb3, const float* __restrict__ cal_slope,
               const float* __restrict__ cal_bias, float* __restrict__ conf_out,
               Ctrl* __restrict__ ctrl) {
  const int row  = blockIdx.x * 8 + (threadIdx.x >> 5);
  const int lane = threadIdx.x & 31;
  float s = 0.f;
  #pragma unroll
  for (int c = lane; c < CH2_; c += 32) s += t2[(size_t)row * CH2_ + c] * cw3[c];
  #pragma unroll
  for (int off = 16; off; off >>= 1) s += __shfl_xor(s, off, 32);
  if (lane == 0) {
    float raw = sigm_f(s + cb3[0]);
    float cf  = sigm_f(cal_slope[0] * (raw - 0.5f) + cal_bias[0]);
    conf_out[row] = cf;
    atomicAdd(&ctrl->conf_sum, cf);
  }
}

__global__ __launch_bounds__(256)
void row_argmax(const float* __restrict__ sim, int ncols,
                int* __restrict__ bidx, float* __restrict__ bsim) {
  __shared__ float sv[256];
  __shared__ int   si[256];
  const int row = blockIdx.x, tid = threadIdx.x;
  float bv = -3.0e38f; int bi = 0x7fffffff;
  for (int c = tid; c < ncols; c += 256) {
    float v = sim[(size_t)row * ncols + c];
    if (v > bv) { bv = v; bi = c; }
  }
  sv[tid] = bv; si[tid] = bi;
  __syncthreads();
  for (int s = 128; s; s >>= 1) {
    if (tid < s) {
      if (sv[tid + s] > sv[tid] ||
          (sv[tid + s] == sv[tid] && si[tid + s] < si[tid])) {
        sv[tid] = sv[tid + s]; si[tid] = si[tid + s];
      }
    }
    __syncthreads();
  }
  if (tid == 0) { bidx[row] = si[0]; bsim[row] = sv[0]; }
}

__global__ __launch_bounds__(256)
void mem_select(const float* __restrict__ state, const float* __restrict__ memv,
                const int* __restrict__ bidx, const float* __restrict__ bsim,
                float* __restrict__ memstate) {
  size_t i = (size_t)blockIdx.x * 256 + threadIdx.x;
  int b = (int)(i / H_), c = (int)(i % H_);
  memstate[i] = (bsim[b] > SIM_TH) ? memv[(size_t)bidx[b] * H_ + c] : state[i];
}

__global__ __launch_bounds__(256)
void commit_state(float* __restrict__ state, const float* __restrict__ nstate,
                  const Ctrl* __restrict__ c) {
  if (!c->do_compute) return;
  size_t i = (size_t)blockIdx.x * 256 + threadIdx.x;
  state[i] = nstate[i];
}

__global__ __launch_bounds__(256)
void commit_conf(float* __restrict__ conf, const float* __restrict__ nconf,
                 const Ctrl* __restrict__ c) {
  if (!c->do_compute) return;
  int i = blockIdx.x * 256 + threadIdx.x;
  conf[i] = nconf[i];
}

__global__ void ctrl_zero(Ctrl* c) { c->conf_sum = 0.f; }

__global__ void ctrl_init(Ctrl* c) {
  float mean = c->conf_sum / (float)B_;
  c->conf_mean = mean;
  float cf = 1.f - mean;
  int amd = MIND + (int)(cf * (float)(MAXD - MIND));
  if (amd > MAXD) amd = MAXD;
  c->amax_depth = amd;
  c->depth = 0;
  c->stopped = 0;
}

__global__ void ctrl_step(Ctrl* c, int d) {
  int step = d + 1;
  int break_now = (step >= MIND) && (c->conf_mean >= CONF_TH);
  int active    = (!c->stopped) && (d < c->amax_depth);
  if (active) c->depth = step;
  c->do_compute   = active && !break_now;
  c->pending_stop = active && break_now;
  c->conf_sum = 0.f;
}

__global__ void ctrl_end(Ctrl* c) {
  if (c->do_compute)   c->conf_mean = c->conf_sum / (float)B_;
  if (c->pending_stop) c->stopped = 1;
}

__global__ __launch_bounds__(256)
void write_out(const float* __restrict__ state, const float* __restrict__ conf,
               const Ctrl* __restrict__ c, float* __restrict__ out) {
  size_t i = (size_t)blockIdx.x * 256 + threadIdx.x;
  const size_t nS = (size_t)B_ * H_;
  if (i < nS)               out[i] = state[i];
  else if (i == nS)         out[i] = (float)c->depth;
  else if (i < nS + 1 + B_) out[i] = conf[i - nS - 1];
}

// ---------------------------------------------------------------------------
extern "C" void kernel_launch(void* const* d_in, const int* in_sizes, int n_in,
                              void* d_out, int out_size, void* d_ws, size_t ws_size,
                              hipStream_t stream) {
  const float* x        = (const float*)d_in[0];
  const float* cw1      = (const float*)d_in[1];
  const float* cb1      = (const float*)d_in[2];
  const float* cw2      = (const float*)d_in[3];
  const float* cb2      = (const float*)d_in[4];
  const float* cw3      = (const float*)d_in[5];
  const float* cb3      = (const float*)d_in[6];
  const float* cal_s    = (const float*)d_in[7];
  const float* cal_b    = (const float*)d_in[8];
  const float* rw1      = (const float*)d_in[9];
  const float* rb1      = (const float*)d_in[10];
  const float* rw2      = (const float*)d_in[11];
  const float* rb2      = (const float*)d_in[12];
  const float* mqw      = (const float*)d_in[13];
  const float* mqb      = (const float*)d_in[14];
  const float* mem_keys = (const float*)d_in[15];
  const float* mem_vals = (const float*)d_in[16];
  const float* mem_use  = (const float*)d_in[17];

  // ---- workspace layout ----
  float* ws = (float*)d_ws;
  const size_t nBH = (size_t)B_ * H_;
  float* state  = ws;                          // [B,H]
  float* nstate = state  + nBH;                // [B,H] (also mem_state)
  float* hbuf   = nstate + nBH;                // [B,2H] (also sim [B,NMEM])
  float* qbuf   = hbuf   + (size_t)B_ * 2 * H_;// [B,H]
  float* keysn  = qbuf   + nBH;                // [NMEM,H]
  float* t1     = keysn  + (size_t)NMEM_ * H_; // [B,CH1]
  float* t2     = t1     + (size_t)B_ * CH1_;  // [B,CH2]
  float* conf   = t2     + (size_t)B_ * CH2_;  // [B]
  float* nconf  = conf   + B_;                 // [B]
  float* bsim   = nconf  + B_;                 // [B]
  int*   bidx   = (int*)(bsim + B_);           // [B]
  Ctrl*  ctrl   = (Ctrl*)(bidx + B_);
  __bf16* packbase = (__bf16*)(((uintptr_t)(ctrl + 1) + 255) & ~(uintptr_t)255);
  __bf16* cw1p  = packbase;                           // H x CH1
  __bf16* cw2p  = cw1p + (size_t)H_ * CH1_;           // CH1 x CH2
  __bf16* rw1p  = cw2p + (size_t)CH1_ * CH2_;         // H x 2H
  __bf16* rw2p  = rw1p + (size_t)H_ * 2 * H_;         // 2H x H
  __bf16* mqwp  = rw2p + (size_t)2 * H_ * H_;         // H x H
  __bf16* keysp = mqwp + (size_t)H_ * H_;             // H x NMEM (keys_n^T)

  const dim3 blk(256);
  const dim3 gBH(32768);  // B*H / 256
  auto pgrid = [](size_t elems) { return dim3((unsigned)(elems / 8 / 256)); };

  // state = x
  hipMemcpyAsync(state, x, nBH * sizeof(float), hipMemcpyDeviceToDevice, stream);
  // keys_n = normalize(mem_keys)
  normalize_rows<<<NMEM_, blk, 0, stream>>>(mem_keys, keysn, H_);

  // ---- pre-pack all B operands into bf16 fragment order ----
  pack_b<false><<<pgrid((size_t)H_ * CH1_),   blk, 0, stream>>>(cw1, cw1p, CH1_, H_);
  pack_b<false><<<pgrid((size_t)CH1_ * CH2_), blk, 0, stream>>>(cw2, cw2p, CH2_, CH1_);
  pack_b<false><<<pgrid((size_t)H_ * 2 * H_), blk, 0, stream>>>(rw1, rw1p, 2 * H_, H_);
  pack_b<false><<<pgrid((size_t)2 * H_ * H_), blk, 0, stream>>>(rw2, rw2p, H_, 2 * H_);
  pack_b<false><<<pgrid((size_t)H_ * H_),     blk, 0, stream>>>(mqw, mqwp, H_, H_);
  pack_b<true ><<<pgrid((size_t)H_ * NMEM_),  blk, 0, stream>>>(keysn, keysp, NMEM_, H_);

  // ---- initial confidence on x ----
  ctrl_zero<<<1, 1, 0, stream>>>(ctrl);
  gemm_bf16_wmma<ACT_GELU, false><<<dim3(CH1_/BN, B_/BM), blk, 0, stream>>>(
      x, cw1p, cb1, nullptr, t1, B_, CH1_, H_);
  gemm_bf16_wmma<ACT_GELU, false><<<dim3(CH2_/BN, B_/BM), blk, 0, stream>>>(
      t1, cw2p, cb2, nullptr, t2, B_, CH2_, CH1_);
  conf_head<<<B_/8, blk, 0, stream>>>(t2, cw3, cb3, cal_s, cal_b, conf, ctrl);
  ctrl_init<<<1, 1, 0, stream>>>(ctrl);

  for (int d = 0; d < MAXD; ++d) {
    ctrl_step<<<1, 1, 0, stream>>>(ctrl, d);

    const float* srcA = state;
    if (d >= 1) {
      gemm_bf16_wmma<ACT_NONE, false><<<dim3(H_/BN, B_/BM), blk, 0, stream>>>(
          state, mqwp, mqb, nullptr, qbuf, B_, H_, H_);
      normalize_rows<<<B_, blk, 0, stream>>>(qbuf, qbuf, H_);
      gemm_bf16_wmma<ACT_NONE, true><<<dim3(NMEM_/BN, B_/BM), blk, 0, stream>>>(
          qbuf, keysp, nullptr, mem_use, hbuf, B_, NMEM_, H_);
      row_argmax<<<B_, blk, 0, stream>>>(hbuf, NMEM_, bidx, bsim);
      mem_select<<<gBH, blk, 0, stream>>>(state, mem_vals, bidx, bsim, nstate);
      srcA = nstate;
    }

    gemm_bf16_wmma<ACT_GELU, false><<<dim3(2*H_/BN, B_/BM), blk, 0, stream>>>(
        srcA, rw1p, rb1, nullptr, hbuf, B_, 2*H_, H_);
    gemm_bf16_wmma<ACT_NONE, false><<<dim3(H_/BN, B_/BM), blk, 0, stream>>>(
        hbuf, rw2p, rb2, nullptr, nstate, B_, H_, 2*H_);

    gemm_bf16_wmma<ACT_GELU, false><<<dim3(CH1_/BN, B_/BM), blk, 0, stream>>>(
        nstate, cw1p, cb1, nullptr, t1, B_, CH1_, H_);
    gemm_bf16_wmma<ACT_GELU, false><<<dim3(CH2_/BN, B_/BM), blk, 0, stream>>>(
        t1, cw2p, cb2, nullptr, t2, B_, CH2_, CH1_);
    conf_head<<<B_/8, blk, 0, stream>>>(t2, cw3, cb3, cal_s, cal_b, nconf, ctrl);

    commit_state<<<gBH, blk, 0, stream>>>(state, nstate, ctrl);
    commit_conf<<<B_/256, blk, 0, stream>>>(conf, nconf, ctrl);
    ctrl_end<<<1, 1, 0, stream>>>(ctrl);
  }

  const size_t total = nBH + 1 + B_;
  write_out<<<dim3((unsigned)((total + 255) / 256)), blk, 0, stream>>>(
      state, conf, ctrl, (float*)d_out);
}
